// BidirectionalRNNClassifier_8143257994060
// MI455X (gfx1250) — matrix operand order repaired and checked by reading, not compiled, not run on hardware
//
#include <hip/hip_runtime.h>
#include <hip/hip_bf16.h>
#include <math.h>

// ---------------------------------------------------------------------------
// BidirectionalRNNClassifier for MI455X (gfx1250, wave32, WMMA bf16 + TDM)
//   B=64, T=512, I=256, D=1024, O=1
// Phase 1 (full-chip parallel): e' = inp@U/sqrt(I)+b  -> bf16, C-fragment
//   swizzled layout (U fragments staged LDS via tensor_load_to_lds).
// Phase 2 (persistent, 32 WGs): 512 sequential steps of
//   s = erf(e'_t + c/sqrt(D)); c = s@W   with v_wmma_f32_16x16x32_bf16,
//   W (2MB bf16) L2-resident in B-fragment layout, one atomic barrier/step.
// Phase 3: tiny head GEMV.
// ---------------------------------------------------------------------------

typedef __attribute__((ext_vector_type(16))) __bf16 v16bf;
typedef __attribute__((ext_vector_type(8)))  __bf16 v8bf;
typedef __attribute__((ext_vector_type(8)))  float  v8f;
typedef __attribute__((ext_vector_type(4)))  float  v4f;
typedef __attribute__((ext_vector_type(4)))  unsigned int u32x4;
typedef __attribute__((ext_vector_type(8)))  int i32x8;
typedef __attribute__((ext_vector_type(4)))  int i32x4;

#define Bm 64
#define Tt 512
#define Ii 256
#define Dd 1024
#define NWG_PER_DIR 16
#define WG_THREADS 256

// workspace byte offsets (all 256B aligned)
#define WS_BAR    0                          // 2 x u32 barrier counters
#define WS_WF     256                        // 1024*1024 bf16 = 2 MiB  (frag W)
#define WS_UF     (WS_WF + 2*1024*1024)      // 256*1024  bf16 = 512 KiB (frag U)
#define WS_SBUF   (WS_UF + 512*1024)         // 2dir*2par*64*1024 bf16 = 512 KiB
#define WS_FINAL  (WS_SBUF + 512*1024)       // 2dir*64*1024 f32 = 512 KiB
#define WS_EBUF   (WS_FINAL + 512*1024)      // 512*65536 bf16 = 64 MiB
#define EBUF_BYTES ((size_t)Tt * 64 * 4 * 256 * 2)

static __device__ inline v16bf cat8(v8bf lo, v8bf hi) {
    return __builtin_shufflevector(lo, hi, 0,1,2,3,4,5,6,7,8,9,10,11,12,13,14,15);
}

// A fragment (16x32 bf16, MxK) from row-major bf16 source.
static __device__ inline v16bf load_a_bf16(const __bf16* src, int ld, int lane) {
    const int kb = (lane & 16) ? 8 : 0;
    const __bf16* p = src + (size_t)(lane & 15) * ld + kb;
    v8bf lo = *(const v8bf*)(p);
    v8bf hi = *(const v8bf*)(p + 16);
    return cat8(lo, hi);
}

// A fragment from row-major f32 source, converted to bf16 on the fly.
static __device__ inline v16bf load_a_f32(const float* src, int ld, int lane) {
    const int kb = (lane & 16) ? 8 : 0;
    const float* p = src + (size_t)(lane & 15) * ld + kb;
    v4f a0 = *(const v4f*)(p);
    v4f a1 = *(const v4f*)(p + 4);
    v4f a2 = *(const v4f*)(p + 16);
    v4f a3 = *(const v4f*)(p + 20);
    v16bf r;
#pragma unroll
    for (int j = 0; j < 4; ++j) {
        r[j]      = (__bf16)a0[j];
        r[4 + j]  = (__bf16)a1[j];
        r[8 + j]  = (__bf16)a2[j];
        r[12 + j] = (__bf16)a3[j];
    }
    return r;
}

// B fragment (32x16 bf16, KxN): each lane's 16 bf16 stored contiguously.
static __device__ inline v16bf load_b_frag(const __bf16* frag, int lane) {
    const __bf16* p = frag + lane * 16;
    v8bf lo = *(const v8bf*)(p);
    v8bf hi = *(const v8bf*)(p + 8);
    return cat8(lo, hi);
}

static __device__ inline v8f wmma_bf16(v16bf a, v16bf b, v8f c) {
    return __builtin_amdgcn_wmma_f32_16x16x32_bf16(
        false, a, false, b, (short)0, c, false, false);
}

// ---------------------------------------------------------------------------
__global__ void init_kernel(unsigned* __restrict__ bar) {
    if (threadIdx.x < 2) bar[threadIdx.x] = 0u;
}

// ---------------------------------------------------------------------------
// Swizzle W (DxD) and U (IxD) f32 -> bf16 WMMA B-fragment layout.
// frag(nglob, kk): K=kk*32+8*(lane>=16)+(j&7)+((j&8)?16:0),
// col = nglob*16+(lane&15); 16 bf16 contiguous per lane.
// ---------------------------------------------------------------------------
__global__ __launch_bounds__(256)
void pack_weights_kernel(const float* __restrict__ W, const float* __restrict__ U,
                         __bf16* __restrict__ Wf, __bf16* __restrict__ Uf) {
    const size_t id = (size_t)blockIdx.x * blockDim.x + threadIdx.x;
    const size_t NW = (size_t)64 * 32 * 512;
    const size_t NU = (size_t)64 * 8 * 512;
    if (id < NW) {
        const int within = (int)(id & 511);
        const int frag   = (int)(id >> 9);
        const int lane = within >> 4, j = within & 15;
        const int nglob = frag >> 5, kk = frag & 31;
        const int k   = kk * 32 + ((lane & 16) ? 8 : 0) + (j & 7) + ((j & 8) ? 16 : 0);
        const int col = nglob * 16 + (lane & 15);
        Wf[id] = (__bf16)W[(size_t)k * Dd + col];
    } else if (id < NW + NU) {
        const size_t id2 = id - NW;
        const int within = (int)(id2 & 511);
        const int frag   = (int)(id2 >> 9);
        const int lane = within >> 4, j = within & 15;
        const int nglob = frag >> 3, kk = frag & 7;
        const int k   = kk * 32 + ((lane & 16) ? 8 : 0) + (j & 7) + ((j & 8) ? 16 : 0);
        const int col = nglob * 16 + (lane & 15);
        Uf[id2] = (__bf16)U[(size_t)k * Dd + col];
    }
}

// ---------------------------------------------------------------------------
// Phase 1: full-chip embed. Grid = T*16 WGs; WG (t, ntile) computes the
// 64x64 tile e'[:, t, ntile] = inp[:,t,:]@U[:,tile]/sqrt(I)+b, stores bf16 in
// C-fragment order: ebuf[((t*64+nglob)*4+mt)*256 + lane*8 + r].
// U fragment slice (32KB) staged global->LDS with the Tensor Data Mover.
// ---------------------------------------------------------------------------
__global__ __launch_bounds__(WG_THREADS)
void embed_kernel(const float* __restrict__ inp, const float* __restrict__ bias,
                  const __bf16* __restrict__ Uf, __bf16* __restrict__ ebuf) {
    __shared__ __bf16 uslice[4 * 8 * 512];   // 32 KiB: this ntile's U fragments

    const int t     = blockIdx.x >> 4;
    const int ntile = blockIdx.x & 15;
    const int tid   = threadIdx.x;
    const int lane  = tid & 31;
    const int wave  = tid >> 5;
    const int mt    = wave & 3;
    const int npair = (wave >> 2) * 2;
    const int m0    = mt * 16;

    const __bf16* uSrc = Uf + (size_t)(ntile * 4) * 8 * 512;   // 8192 dwords

    if (wave == 0) {
        // Build Tensor DMA Descriptor: 1-D tile of 8192 dwords, global->LDS.
        unsigned long long ga = (unsigned long long)(size_t)uSrc;
        unsigned lds = (unsigned)(size_t)&uslice[0];   // low 32 bits = LDS offset
        u32x4 g0 = {};
        g0[0] = 1u;                                    // count=1 (valid), user mode
        g0[1] = lds;                                   // lds_addr
        g0[2] = (unsigned)(ga & 0xffffffffu);          // global_addr[31:0]
        g0[3] = (unsigned)((ga >> 32) & 0x01ffffffu) | (2u << 30);  // addr[56:32], type=2
        i32x8 g1 = {};
        g1[0] = (2 << 16);                 // data_size = 4B dwords
        g1[1] = (int)((8192u & 0xffffu) << 16);        // tensor_dim0 lo16
        g1[2] = (int)((8192u >> 16) | (1u << 16));     // tensor_dim0 hi16, tensor_dim1=1
        g1[3] = (int)(8192u << 16);                    // tile_dim0 = 8192
        g1[4] = 1;                                     // tile_dim1 = 1
        g1[5] = 8192;                                  // tensor_dim0_stride
        g1[6] = (int)((8192u & 0xffffu) << 16);        // tensor_dim1_stride lo16
        g1[7] = 0;
        i32x4 gz = {};
#if defined(__clang_major__) && (__clang_major__ >= 23)
        i32x8 gz8 = {};
        __builtin_amdgcn_tensor_load_to_lds(g0, g1, gz, gz, gz8, 0);
#else
        __builtin_amdgcn_tensor_load_to_lds(g0, g1, gz, gz, 0);
#endif
        __builtin_amdgcn_s_wait_tensorcnt(0);
    }
    __syncthreads();

    const float invSqrtI = 0.0625f;
    const int ng0 = ntile * 4 + npair, ng1 = ng0 + 1;
    const float bias0 = bias[ng0 * 16 + (lane & 15)];
    const float bias1 = bias[ng1 * 16 + (lane & 15)];

    v8f e0 = {}, e1 = {};
    const float* aBase = inp + (size_t)m0 * (Tt * Ii) + (size_t)t * Ii;
#pragma unroll
    for (int kk = 0; kk < Ii / 32; ++kk) {
        v16bf a  = load_a_f32(aBase + kk * 32, Tt * Ii, lane);
        v16bf b0 = load_b_frag(uslice + ((npair * 8 + kk) * 512), lane);        // ds_load_b128
        v16bf b1 = load_b_frag(uslice + (((npair + 1) * 8 + kk) * 512), lane);
        e0 = wmma_bf16(a, b0, e0);
        e1 = wmma_bf16(a, b1, e1);
    }

    v8bf p0, p1;
#pragma unroll
    for (int r = 0; r < 8; ++r) {
        p0[r] = (__bf16)(e0[r] * invSqrtI + bias0);
        p1[r] = (__bf16)(e1[r] * invSqrtI + bias1);
    }
    *(v8bf*)(ebuf + ((size_t)((t * 64 + ng0) * 4 + mt)) * 256 + lane * 8) = p0;
    *(v8bf*)(ebuf + ((size_t)((t * 64 + ng1) * 4 + mt)) * 256 + lane * 8) = p1;
}

// ---------------------------------------------------------------------------
// Phase 2: persistent recurrence. 32 WGs (16/dir). Each wave: two 16x16 tiles.
// use_ebuf=1: read precomputed e' fragments (1 b128 load/tile).
// use_ebuf=0: fallback, fused embed GEMM from inp/Uf.
// ---------------------------------------------------------------------------
__global__ __launch_bounds__(WG_THREADS)
void rnn_recurrence_kernel(const float* __restrict__ inp,
                           const float* __restrict__ bias,
                           const __bf16* __restrict__ Wf,
                           const __bf16* __restrict__ Uf,
                           const __bf16* __restrict__ ebuf, int use_ebuf,
                           __bf16* __restrict__ sbuf,
                           float* __restrict__ finalbuf,
                           unsigned* __restrict__ bar) {
    const int dir   = blockIdx.x >> 4;
    const int ntile = blockIdx.x & 15;
    const int tid   = threadIdx.x;
    const int lane  = tid & 31;
    const int wave  = tid >> 5;
    const int mt    = wave & 3;
    const int npair = (wave >> 2) * 2;
    const int m0    = mt * 16;

    const float invSqrtI = 0.0625f;
    const float invSqrtD = 0.03125f;
    const int ng0 = ntile * 4 + npair, ng1 = ng0 + 1;
    const float bias0 = bias[ng0 * 16 + (lane & 15)];
    const float bias1 = bias[ng1 * 16 + (lane & 15)];

    unsigned* mybar = bar + dir;
    __bf16* sb0 = sbuf + (size_t)(dir * 2 + 0) * Bm * Dd;
    __bf16* sb1 = sbuf + (size_t)(dir * 2 + 1) * Bm * Dd;

    // Warm L2 with this WG's W fragment slice (128 KiB, reused 511x).
    {
        const char* base = (const char*)(Wf + (size_t)(ntile * 4) * 32 * 512);
        for (int o = tid * 128; o < 128 * 1024; o += WG_THREADS * 128)
            __builtin_prefetch(base + o, 0, 1);
    }

    const v8f vzero = {};
    v8f cAcc0 = vzero, cAcc1 = vzero;

    for (int t = 0; t < Tt; ++t) {
        const int tcur = dir ? (Tt - 1 - t) : t;

        // ---- obtain e' tile values (scale+bias already folded when use_ebuf) ----
        float ev0[8], ev1[8];
        if (use_ebuf) {
            v8bf f0 = *(const v8bf*)(ebuf + ((size_t)((tcur * 64 + ng0) * 4 + mt)) * 256 + lane * 8);
            v8bf f1 = *(const v8bf*)(ebuf + ((size_t)((tcur * 64 + ng1) * 4 + mt)) * 256 + lane * 8);
#pragma unroll
            for (int r = 0; r < 8; ++r) { ev0[r] = (float)f0[r]; ev1[r] = (float)f1[r]; }
        } else {
            v8f e0 = vzero, e1 = vzero;
            const float* aBase = inp + (size_t)m0 * (Tt * Ii) + (size_t)tcur * Ii;
#pragma unroll
            for (int kk = 0; kk < Ii / 32; ++kk) {
                v16bf a  = load_a_f32(aBase + kk * 32, Tt * Ii, lane);
                v16bf b0 = load_b_frag(Uf + ((size_t)ng0 * 8 + kk) * 512, lane);
                v16bf b1 = load_b_frag(Uf + ((size_t)ng1 * 8 + kk) * 512, lane);
                e0 = wmma_bf16(a, b0, e0);
                e1 = wmma_bf16(a, b1, e1);
            }
#pragma unroll
            for (int r = 0; r < 8; ++r) {
                ev0[r] = e0[r] * invSqrtI + bias0;
                ev1[r] = e1[r] * invSqrtI + bias1;
            }
        }

        // ---- s = erf(e' + c/sqrt(D)) ----
        const int rowBase = m0 + ((lane & 16) ? 8 : 0);
        const int colg0 = ng0 * 16 + (lane & 15);
        const int colg1 = ng1 * 16 + (lane & 15);
        const int par = t & 1;

        if (t == Tt - 1) {
            float* f = finalbuf + (size_t)dir * Bm * Dd;
#pragma unroll
            for (int r = 0; r < 8; ++r) {
                f[(size_t)(rowBase + r) * Dd + colg0] = erff(ev0[r] + cAcc0[r] * invSqrtD);
                f[(size_t)(rowBase + r) * Dd + colg1] = erff(ev1[r] + cAcc1[r] * invSqrtD);
            }
            break;
        }

        __bf16* s = par ? sb1 : sb0;
#pragma unroll
        for (int r = 0; r < 8; ++r) {
            s[(size_t)(rowBase + r) * Dd + colg0] = (__bf16)erff(ev0[r] + cAcc0[r] * invSqrtD);
            s[(size_t)(rowBase + r) * Dd + colg1] = (__bf16)erff(ev1[r] + cAcc1[r] * invSqrtD);
        }

        // ---- one grid barrier per step over the 16 WGs of this direction ----
        __threadfence();
        __syncthreads();
        if (tid == 0) {
            __hip_atomic_fetch_add(mybar, 1u, __ATOMIC_RELEASE, __HIP_MEMORY_SCOPE_AGENT);
            const unsigned target = (unsigned)(t + 1) * NWG_PER_DIR;
            while (__hip_atomic_load(mybar, __ATOMIC_ACQUIRE, __HIP_MEMORY_SCOPE_AGENT) < target)
                __builtin_amdgcn_s_sleep(1);
        }
        __syncthreads();

        // ---- carry GEMM: c = s(64x1024) @ W[:,tile], bf16 WMMA, L2-hot ----
        cAcc0 = vzero; cAcc1 = vzero;
        const __bf16* sA = (par ? sb1 : sb0) + (size_t)m0 * Dd;
#pragma unroll 4
        for (int kk = 0; kk < Dd / 32; ++kk) {
            v16bf a  = load_a_bf16(sA + kk * 32, Dd, lane);
            v16bf b0 = load_b_frag(Wf + ((size_t)ng0 * 32 + kk) * 512, lane);
            v16bf b1 = load_b_frag(Wf + ((size_t)ng1 * 32 + kk) * 512, lane);
            cAcc0 = wmma_bf16(a, b0, cAcc0);
            cAcc1 = wmma_bf16(a, b1, cAcc1);
        }
    }
}

// ---------------------------------------------------------------------------
__global__ __launch_bounds__(256)
void head_kernel(const float* __restrict__ finalbuf,
                 const float* __restrict__ v,
                 float* __restrict__ out) {
    __shared__ float red[256];
    const int b = blockIdx.x;
    const int tid = threadIdx.x;
    const float* sf  = finalbuf + (size_t)b * Dd;
    const float* sbk = finalbuf + (size_t)Bm * Dd + (size_t)b * Dd;
    float acc = 0.f;
    for (int d = tid; d < Dd; d += 256)
        acc += sf[d] * v[d] + sbk[d] * v[Dd + d];
    red[tid] = acc;
    __syncthreads();
    for (int sft = 128; sft > 0; sft >>= 1) {
        if (tid < sft) red[tid] += red[tid + sft];
        __syncthreads();
    }
    if (tid == 0) out[b] = red[0] * 0.03125f;
}

// ---------------------------------------------------------------------------
extern "C" void kernel_launch(void* const* d_in, const int* in_sizes, int n_in,
                              void* d_out, int out_size, void* d_ws, size_t ws_size,
                              hipStream_t stream) {
    const float* inp = (const float*)d_in[0];   // (B,T,I)
    const float* W   = (const float*)d_in[1];   // (D,D)
    const float* U   = (const float*)d_in[2];   // (I,D)
    const float* bia = (const float*)d_in[3];   // (D,)
    const float* v   = (const float*)d_in[4];   // (2D,O)
    float* out = (float*)d_out;

    char* ws = (char*)d_ws;
    unsigned* bar   = (unsigned*)(ws + WS_BAR);
    __bf16* Wf      = (__bf16*)(ws + WS_WF);
    __bf16* Uf      = (__bf16*)(ws + WS_UF);
    __bf16* sbuf    = (__bf16*)(ws + WS_SBUF);
    float* finalbuf = (float*)(ws + WS_FINAL);
    __bf16* ebuf    = (__bf16*)(ws + WS_EBUF);

    const int use_ebuf = (ws_size >= (size_t)WS_EBUF + EBUF_BYTES) ? 1 : 0;

    hipLaunchKernelGGL(init_kernel, dim3(1), dim3(32), 0, stream, bar);

    const int packElems = 1048576 + 262144;
    hipLaunchKernelGGL(pack_weights_kernel, dim3((packElems + 255) / 256), dim3(256),
                       0, stream, W, U, Wf, Uf);

    if (use_ebuf) {
        hipLaunchKernelGGL(embed_kernel, dim3(Tt * 16), dim3(WG_THREADS), 0, stream,
                           inp, bia, Uf, ebuf);
    }

    hipLaunchKernelGGL(rnn_recurrence_kernel, dim3(2 * NWG_PER_DIR), dim3(WG_THREADS),
                       0, stream, inp, bia, Wf, Uf, ebuf, use_ebuf, sbuf, finalbuf, bar);

    hipLaunchKernelGGL(head_kernel, dim3(Bm), dim3(256), 0, stream, finalbuf, v, out);
}
